// Yamoe_89120571392709
// MI455X (gfx1250) — matrix operand order repaired and checked by reading, not compiled, not run on hardware
//
#include <hip/hip_runtime.h>
#include <stdint.h>

#define T_TOK 4096
#define HDIM  1024
#define NEXP  32
#define TOPK  4
#define CAP   1024
#define ALPHA 1.702f
#define LIMIT 7.0f
#define NKSTEP (HDIM / 32)   // 32 K-steps of 32

typedef __attribute__((ext_vector_type(16))) __bf16 v16bf;
typedef __attribute__((ext_vector_type(8)))  float  v8f;

__device__ __forceinline__ unsigned pack2_bf16(float a, float b) {
    // round-to-nearest-even f32 -> bf16, packed (a in [15:0], b in [31:16])
    unsigned ua = __builtin_bit_cast(unsigned, a);
    unsigned ub = __builtin_bit_cast(unsigned, b);
    ua = ua + 0x7fffu + ((ua >> 16) & 1u);
    ub = ub + 0x7fffu + ((ub >> 16) & 1u);
    return (ua >> 16) | (ub & 0xffff0000u);
}
__device__ __forceinline__ unsigned short f32_to_bf16(float a) {
    unsigned ua = __builtin_bit_cast(unsigned, a);
    ua = ua + 0x7fffu + ((ua >> 16) & 1u);
    return (unsigned short)(ua >> 16);
}

// CDNA5 async copy: 16B global -> LDS, tracked by ASYNCcnt.
__device__ __forceinline__ void async_cp16(unsigned lds_off, const void* gsrc) {
    asm volatile("global_load_async_to_lds_b128 %0, %1, off"
                 :: "v"(lds_off), "v"(gsrc) : "memory");
}
__device__ __forceinline__ void wait_async0() {
    asm volatile("s_wait_asynccnt 0" ::: "memory");
}

// ---------------------------------------------------------------------------
// Router: logits[t,e] = x[t,:] . rw[e,:] + rb[e]   (one block per token)
// ---------------------------------------------------------------------------
__global__ __launch_bounds__(256) void router_kernel(
    const float* __restrict__ x, const float* __restrict__ rw,
    const float* __restrict__ rb, float* __restrict__ logits) {
    const int t = blockIdx.x;
    __shared__ float xs[HDIM];
    __shared__ float part[8][NEXP];
    const float* xp = x + (size_t)t * HDIM;
    for (int i = threadIdx.x; i < HDIM; i += 256) xs[i] = xp[i];
    __syncthreads();
    const int e = threadIdx.x & 31;
    const int c = threadIdx.x >> 5;
    const float* wrow = rw + (size_t)e * HDIM + c * 128;
    float s = 0.f;
    #pragma unroll 8
    for (int j = 0; j < 128; ++j) s += xs[c * 128 + j] * wrow[j];
    part[c][e] = s;
    __syncthreads();
    if (threadIdx.x < NEXP) {
        float tot = rb[threadIdx.x];
        #pragma unroll
        for (int cc = 0; cc < 8; ++cc) tot += part[cc][threadIdx.x];
        logits[(size_t)t * NEXP + threadIdx.x] = tot;
    }
}

// ---------------------------------------------------------------------------
// Top-4 + softmax; writes dense router_scores and (idx, weight) per (t,k)
// ---------------------------------------------------------------------------
__global__ __launch_bounds__(256) void topk_kernel(
    const float* __restrict__ logits, float* __restrict__ scores,
    int* __restrict__ top_idx, float* __restrict__ w_arr) {
    const int t = blockIdx.x * 256 + threadIdx.x;
    if (t >= T_TOK) return;
    float lg[NEXP];
    #pragma unroll
    for (int e = 0; e < NEXP; ++e) lg[e] = logits[(size_t)t * NEXP + e];
    float vals[TOPK]; int idx[TOPK];
    unsigned chosen = 0u;
    #pragma unroll
    for (int k = 0; k < TOPK; ++k) {
        float bv = -__builtin_inff(); int bi = 0;
        #pragma unroll
        for (int e = 0; e < NEXP; ++e)
            if (!((chosen >> e) & 1u) && lg[e] > bv) { bv = lg[e]; bi = e; }
        vals[k] = bv; idx[k] = bi; chosen |= (1u << bi);
    }
    const float m = vals[0];
    float ex[TOPK], ssum = 0.f;
    #pragma unroll
    for (int k = 0; k < TOPK; ++k) { ex[k] = __expf(vals[k] - m); ssum += ex[k]; }
    const float inv = 1.f / ssum;
    #pragma unroll
    for (int e = 0; e < NEXP; ++e) {
        float v = 0.f;
        #pragma unroll
        for (int k = 0; k < TOPK; ++k) if (idx[k] == e) v = ex[k] * inv;
        scores[(size_t)t * NEXP + e] = v;
    }
    #pragma unroll
    for (int k = 0; k < TOPK; ++k) {
        top_idx[t * TOPK + k] = idx[k];
        w_arr[t * TOPK + k]   = ex[k] * inv;
    }
}

// ---------------------------------------------------------------------------
// Per-expert sequential rank over flat (t*K+k) order; slot=-1 if over capacity
// One block per expert; ballot-based block prefix scan, wave32.
// ---------------------------------------------------------------------------
__global__ __launch_bounds__(256) void scan_kernel(
    const int* __restrict__ top_idx, int* __restrict__ slot_arr) {
    const int b = blockIdx.x;
    __shared__ int wsum[8];
    const int lane = threadIdx.x & 31;
    const int wid  = threadIdx.x >> 5;
    int base = 0;
    for (int c = 0; c < (T_TOK * TOPK) / 256; ++c) {
        const int i = c * 256 + threadIdx.x;
        const bool flag = (top_idx[i] == b);
        const unsigned m32 = (unsigned)__ballot(flag);
        const int prefix = __popc(m32 & ((1u << lane) - 1u));
        if (lane == 0) wsum[wid] = __popc(m32);
        __syncthreads();
        int woff = 0, tot = 0;
        #pragma unroll
        for (int w = 0; w < 8; ++w) { if (w < wid) woff += wsum[w]; tot += wsum[w]; }
        if (flag) {
            const int rank = base + woff + prefix;
            slot_arr[i] = (rank < CAP) ? rank : -1;
        }
        base += tot;
        __syncthreads();
    }
}

// ---------------------------------------------------------------------------
// Dispatch: bufA[e, slot, :] = bf16(x[t, :])   (row-major bf16, dword = K-pair)
// ---------------------------------------------------------------------------
__global__ __launch_bounds__(256) void dispatch_kernel(
    const float* __restrict__ x, const int* __restrict__ top_idx,
    const int* __restrict__ slot_arr, unsigned* __restrict__ bufA) {
    const int i = blockIdx.x;
    const int s = slot_arr[i];
    if (s < 0) return;
    const int e = top_idx[i];
    const int t = i >> 2;
    const float2* src = (const float2*)(x + (size_t)t * HDIM);
    unsigned* dst = bufA + ((size_t)e * CAP + s) * (HDIM / 2);
    #pragma unroll
    for (int j = 0; j < 2; ++j) {
        const int c = threadIdx.x + j * 256;   // 512 dwords per row
        const float2 v = src[c];
        dst[c] = pack2_bf16(v.x, v.y);
    }
}

// ---------------------------------------------------------------------------
// Weight pre-convert/swizzle: f32 [H, NB] -> bf16 tiles laid out exactly as
// the WMMA B-fragment LDS image: tile(e, nt, ks) = 128 cols x 16 K-pair
// dwords, contiguous 8KB. Enables pure async b128 staging in the GEMM.
// ---------------------------------------------------------------------------
__global__ __launch_bounds__(256) void convert_w_kernel(
    const float* __restrict__ W, unsigned* __restrict__ out, int NB) {
    const int ks = blockIdx.x;           // K step (32 K values)
    const int nt = blockIdx.y;           // N tile (128 cols)
    const int e  = blockIdx.z;
    const float* Wb = W + (size_t)e * HDIM * NB;
    unsigned* tile =
        out + (((size_t)e * gridDim.y + nt) * NKSTEP + ks) * 2048;
    const int k0 = ks * 32, n0 = nt * 128;
    #pragma unroll
    for (int i = 0; i < 8; ++i) {
        const int idx = threadIdx.x + i * 256;
        const int n = idx & 127, v = idx >> 7;
        const float* s = Wb + (size_t)(k0 + 2 * v) * NB + (n0 + n);
        tile[n * 16 + v] = pack2_bf16(s[0], s[NB]);
    }
}

// ---------------------------------------------------------------------------
// Tiled WMMA GEMM, bf16 x bf16 -> f32 accum. A: bf16 row-major [CAP, H].
// Bsw: pre-swizzled bf16 tiles. Async global->LDS staging, double-buffered
// via a runtime ping-pong offset (loop kept rolled so accumulators stay
// in-place across iterations: no v_mov shuffles / hazard nops between WMMAs).
// Block tile 128x128, 8 waves (2x4), wave tile 64x32, K-step 32.
// ---------------------------------------------------------------------------
template <bool GLU>
__global__ __launch_bounds__(256) void gemm_kernel(
    const unsigned short* __restrict__ Abase, const unsigned* __restrict__ Bsw,
    const float* __restrict__ biasBase, void* __restrict__ Obase, int NB) {
    const int e   = blockIdx.z;
    const int m0  = blockIdx.y * 128;
    const int nt  = blockIdx.x;
    const int n0  = nt * 128;
    const int tid = threadIdx.x;
    const int lane = tid & 31;
    const int wid  = tid >> 5;
    const int wm = wid & 1;   // 2 waves over M (64 rows each)
    const int wn = wid >> 1;  // 4 waves over N (32 cols each)
    const int hl  = (lane < 16) ? 0 : 1;
    const int l15 = lane & 15;

    const unsigned short* A = Abase + (size_t)e * CAP * HDIM;
    const unsigned* Btiles = Bsw + (((size_t)e * gridDim.x + nt) * NKSTEP) * 2048;

    // bf16 pairs; rows padded to 20 dwords (16B aligned, bank-spread)
    __shared__ unsigned lds_a[2][128][20];  // [buf][m][pair p]
    __shared__ unsigned lds_b[2][128][20];  // [buf][n][pair v]
    const unsigned BUFB = sizeof(lds_a[0]);       // 10240 bytes
    const unsigned BUFD = BUFB / 4;               // 2560 dwords

    // per-thread async staging addresses (buffer 0 base; buffer 1 = +BUFB)
    unsigned a_dst[2], b_dst[2];
    const unsigned short* a_src[2];
    const unsigned* b_src[2];
    #pragma unroll
    for (int j = 0; j < 2; ++j) {
        const int c = tid + j * 256;              // 0..511 16B-chunks
        const int m = c >> 2, q = c & 3;
        a_dst[j] = (unsigned)(size_t)&lds_a[0][m][q * 4];
        a_src[j] = A + (size_t)(m0 + m) * HDIM + q * 8;
        b_dst[j] = (unsigned)(size_t)&lds_b[0][m][q * 4];  // same c mapping
        b_src[j] = Btiles + c * 4;
    }
    // fragment read base pointers (buffer 0; +BUFD dwords for buffer 1)
    const unsigned* fa = &lds_a[0][wm * 64 + l15][hl * 4];
    const unsigned* fb = &lds_b[0][wn * 32 + l15][hl * 4];

    v8f acc[4][2];
    #pragma unroll
    for (int i = 0; i < 4; ++i)
        #pragma unroll
        for (int j = 0; j < 2; ++j)
            acc[i][j] = (v8f){0.f, 0.f, 0.f, 0.f, 0.f, 0.f, 0.f, 0.f};

    // prologue: stage K-step 0 into buffer 0
    #pragma unroll
    for (int j = 0; j < 2; ++j) {
        async_cp16(a_dst[j], a_src[j]);
        async_cp16(b_dst[j], b_src[j]);
    }

    unsigned bod = 0;     // dword offset of current buffer (0 or BUFD)
    #pragma unroll 1
    for (int ks = 0; ks < NKSTEP; ++ks) {
        wait_async0();        // own copies for current buffer complete
        __syncthreads();      // all waves' copies visible; prev readers done

        // issue next K-step into the other buffer first: overlaps with the
        // fragment ds_loads and the 8 WMMAs below.
        if (ks + 1 < NKSTEP) {
            const unsigned nbo = (bod == 0) ? BUFB : 0u;  // next buffer (bytes)
            #pragma unroll
            for (int j = 0; j < 2; ++j) {
                async_cp16(a_dst[j] + nbo, a_src[j] + (size_t)(ks + 1) * 32);
                async_cp16(b_dst[j] + nbo, b_src[j] + (size_t)(ks + 1) * 2048);
            }
        }

        union Frag { uint4 q[2]; v16bf v; } afr[4], bfr[2];
        #pragma unroll
        for (int tn = 0; tn < 2; ++tn) {
            const unsigned* bp = fb + bod + (size_t)tn * 16 * 20;
            bfr[tn].q[0] = *(const uint4*)(bp);
            bfr[tn].q[1] = *(const uint4*)(bp + 8);
        }
        #pragma unroll
        for (int tm = 0; tm < 4; ++tm) {
            const unsigned* ap = fa + bod + (size_t)tm * 16 * 20;
            afr[tm].q[0] = *(const uint4*)(ap);
            afr[tm].q[1] = *(const uint4*)(ap + 8);
        }

        #pragma unroll
        for (int tm = 0; tm < 4; ++tm)
            #pragma unroll
            for (int tn = 0; tn < 2; ++tn)
                acc[tm][tn] = __builtin_amdgcn_wmma_f32_16x16x32_bf16(
                    false, afr[tm].v, false, bfr[tn].v,
                    (short)0, acc[tm][tn], false, false);

        bod ^= BUFD;   // value toggle (0 <-> BUFD), applied via addition above
    }

    // epilogue
    #pragma unroll
    for (int tn = 0; tn < 2; ++tn) {
        const int col = n0 + wn * 32 + tn * 16 + l15;
        const float bias = biasBase[(size_t)e * NB + col];
        #pragma unroll
        for (int tm = 0; tm < 4; ++tm) {
            const int rbase = m0 + wm * 64 + tm * 16 + hl * 8;
            #pragma unroll
            for (int r = 0; r < 8; ++r) {
                float v = acc[tm][tn][r] + bias;
                if (GLU) {
                    const float partner = __shfl_xor(v, 1, 32);
                    if ((l15 & 1) == 0) {
                        float gate = fminf(v, LIMIT);
                        float up   = fminf(fmaxf(partner, -LIMIT), LIMIT);
                        float glu  = gate / (1.0f + __expf(-ALPHA * gate));
                        ((unsigned short*)Obase)[(size_t)e * CAP * (NB / 2) +
                            (size_t)(rbase + r) * (NB / 2) + (col >> 1)] =
                            f32_to_bf16((up + 1.0f) * glu);
                    }
                } else {
                    ((float*)Obase)[(size_t)e * CAP * NB +
                                    (size_t)(rbase + r) * NB + col] = v;
                }
            }
        }
    }
}

// ---------------------------------------------------------------------------
// Gather: out[t,:] = sum_k w[t,k] * expert_out[e(t,k), slot(t,k), :]
// ---------------------------------------------------------------------------
__global__ __launch_bounds__(256) void gather_kernel(
    const float* __restrict__ eout, const int* __restrict__ top_idx,
    const int* __restrict__ slot_arr, const float* __restrict__ w_arr,
    float* __restrict__ out) {
    const int t = blockIdx.x;
    const int c = threadIdx.x;  // float4 column
    float4 acc = {0.f, 0.f, 0.f, 0.f};
    #pragma unroll
    for (int k = 0; k < TOPK; ++k) {
        const int i = t * TOPK + k;
        const int s = slot_arr[i];
        if (s >= 0) {
            const int e = top_idx[i];
            const float w = w_arr[i];
            const float4 v =
                ((const float4*)(eout + ((size_t)e * CAP + s) * HDIM))[c];
            acc.x += w * v.x; acc.y += w * v.y;
            acc.z += w * v.z; acc.w += w * v.w;
        }
    }
    ((float4*)(out + (size_t)t * HDIM))[c] = acc;
}

// ---------------------------------------------------------------------------
extern "C" void kernel_launch(void* const* d_in, const int* in_sizes, int n_in,
                              void* d_out, int out_size, void* d_ws, size_t ws_size,
                              hipStream_t stream) {
    (void)in_sizes; (void)n_in; (void)out_size; (void)ws_size;
    const float* x     = (const float*)d_in[0];  // [2,2048,1024]
    const float* rw    = (const float*)d_in[1];  // [32,1024]
    const float* rb    = (const float*)d_in[2];  // [32]
    const float* gup   = (const float*)d_in[3];  // [32,1024,2048]
    const float* gup_b = (const float*)d_in[4];  // [32,2048]
    const float* dp    = (const float*)d_in[5];  // [32,1024,1024]
    const float* dp_b  = (const float*)d_in[6];  // [32,1024]

    float* out    = (float*)d_out;                   // [4096*1024]
    float* scores = out + (size_t)T_TOK * HDIM;      // [4096*32]

    char* ws = (char*)d_ws;
    float* logits   = (float*)(ws);                  // 4096*32 f32
    int*   top_idx  = (int*)  (ws + 524288);
    float* w_arr    = (float*)(ws + 589824);
    int*   slot_arr = (int*)  (ws + 655360);
    unsigned short* bufA = (unsigned short*)(ws + (1 << 20));        // 64MB bf16 [E,CAP,H]
    unsigned short* actb = bufA + (size_t)NEXP * CAP * HDIM;         // 64MB bf16 [E,CAP,H]
    unsigned* gupw = (unsigned*)(actb + (size_t)NEXP * CAP * HDIM);  // 128MB swizzled bf16
    unsigned* dpw  = gupw + (size_t)NEXP * HDIM * (2 * HDIM) / 2;    // 64MB swizzled bf16
    float* eout = (float*)gupw;  // 128MB f32 [E,CAP,H]; aliases gupw (dead after GEMM1)

    hipMemsetAsync(bufA, 0, (size_t)NEXP * CAP * HDIM * sizeof(unsigned short), stream);

    router_kernel<<<T_TOK, 256, 0, stream>>>(x, rw, rb, logits);
    topk_kernel<<<T_TOK / 256, 256, 0, stream>>>(logits, scores, top_idx, w_arr);
    scan_kernel<<<NEXP, 256, 0, stream>>>(top_idx, slot_arr);
    dispatch_kernel<<<T_TOK * TOPK, 256, 0, stream>>>(x, top_idx, slot_arr,
                                                      (unsigned*)bufA);
    convert_w_kernel<<<dim3(NKSTEP, 16, NEXP), 256, 0, stream>>>(gup, gupw, 2 * HDIM);
    convert_w_kernel<<<dim3(NKSTEP, 8, NEXP), 256, 0, stream>>>(dp, dpw, HDIM);

    // GEMM1: bf16 [CAP,H] x [H,2H] -> fused GLU -> bf16 act [CAP,H]
    gemm_kernel<true><<<dim3(16, CAP / 128, NEXP), 256, 0, stream>>>(
        bufA, gupw, gup_b, actb, 2 * HDIM);
    // GEMM2: bf16 [CAP,H] x [H,H] -> f32 expert_out
    gemm_kernel<false><<<dim3(8, CAP / 128, NEXP), 256, 0, stream>>>(
        actb, dpw, dp_b, eout, HDIM);

    gather_kernel<<<T_TOK, 256, 0, stream>>>(eout, top_idx, slot_arr, w_arr, out);
}
